// MoELayer_82257213653336
// MI455X (gfx1250) — compile-verified, matrix operand order checked by reading, not run on hardware
//
#include <hip/hip_runtime.h>

// ---------------------------------------------------------------------------
// MoE top-1 routed layer for MI455X (gfx1250), wave32 + WMMA f16->f32.
// B=4, T=2048, D=1024, H=4096, E=8.  BT = 8192 tokens.
//
// gate+argmax -> expert buckets -> grouped GEMM1 (relu(x@W1+b1) -> f16 h)
//             -> grouped GEMM2 (h@W2+b2 -> scatter fp32 by token).
// GEMMs: 128x128x32 tiles, double-buffered LDS, software-pipelined global
// loads, async-to-LDS staging for the f16 h operand (GLOBAL_LOAD_ASYNC_TO_LDS).
// ---------------------------------------------------------------------------

#define BT 8192
#define DD 1024
#define HH 4096
#define EE 8
#define LDSTRIDE 40   // halves per LDS row (32 + 8 pad -> conflict-free b128)

typedef _Float16 v4h  __attribute__((ext_vector_type(4)));
typedef _Float16 v8h  __attribute__((ext_vector_type(8)));
typedef _Float16 v16h __attribute__((ext_vector_type(16)));
typedef float    v8f  __attribute__((ext_vector_type(8)));
typedef float    v4f  __attribute__((ext_vector_type(4)));
typedef int      v4i  __attribute__((ext_vector_type(4)));

#define WS_EXPERT_ID 32
#define WS_TOKLIST   (32 + BT)
#define WS_HBUF_BYTE 65664

#if __has_builtin(__builtin_amdgcn_global_load_async_to_lds_b128)
#define HAVE_ASYNC_LDS 1
#else
#define HAVE_ASYNC_LDS 0
#endif

#if HAVE_ASYNC_LDS
typedef __attribute__((address_space(1))) v4i* g16p;   // global 16B chunk ptr
typedef __attribute__((address_space(3))) v4i* l16p;   // LDS 16B chunk ptr
#endif

__device__ __forceinline__ void async_copy16(const _Float16* g, _Float16* l) {
#if HAVE_ASYNC_LDS
    __builtin_amdgcn_global_load_async_to_lds_b128((g16p)g, (l16p)l, 0, 0);
#else
    *(v8h*)l = *(const v8h*)g;
#endif
}

__device__ __forceinline__ void async_wait() {
#if HAVE_ASYNC_LDS
    asm volatile("s_wait_asynccnt 0x0" ::: "memory");
#endif
}

// ---------------------------------------------------------------------------
__global__ __launch_bounds__(32) void zero_ws_kernel(int* iws) {
    iws[threadIdx.x] = 0;  // counts, cnt2, offsets, spare
}

__global__ __launch_bounds__(256) void gate_kernel(
    const float* __restrict__ x, const float* __restrict__ Wg,
    const float* __restrict__ bg, int* __restrict__ expert_id,
    int* __restrict__ counts) {
    const int lane = threadIdx.x & 31;
    const int wid  = threadIdx.x >> 5;
    const int t    = blockIdx.x * 8 + wid;
    const float* xt = x + (size_t)t * DD;

    float s[EE];
    #pragma unroll
    for (int e = 0; e < EE; ++e) s[e] = 0.f;

    for (int d = lane; d < DD; d += 32) {
        const float xv = xt[d];
        const v4f* w = (const v4f*)(Wg + (size_t)d * EE);
        const v4f w0 = w[0], w1 = w[1];
        s[0] += xv * w0[0]; s[1] += xv * w0[1];
        s[2] += xv * w0[2]; s[3] += xv * w0[3];
        s[4] += xv * w1[0]; s[5] += xv * w1[1];
        s[6] += xv * w1[2]; s[7] += xv * w1[3];
    }
    #pragma unroll
    for (int o = 16; o > 0; o >>= 1)
        #pragma unroll
        for (int e = 0; e < EE; ++e) s[e] += __shfl_xor(s[e], o, 32);

    if (lane == 0) {
        int best = 0;
        float bv = s[0] + bg[0];
        #pragma unroll
        for (int e = 1; e < EE; ++e) {
            const float v = s[e] + bg[e];
            if (v > bv) { bv = v; best = e; }  // first-max wins (argmax)
        }
        expert_id[t] = best;
        atomicAdd(&counts[best], 1);
    }
}

__global__ void offsets_kernel(int* iws) {
    int acc = 0;
    for (int e = 0; e < EE; ++e) {
        iws[16 + e] = acc;
        acc += iws[e];
    }
}

__global__ __launch_bounds__(256) void scatter_kernel(int* iws) {
    const int t = blockIdx.x * 256 + threadIdx.x;
    const int e = iws[WS_EXPERT_ID + t];
    const int p = atomicAdd(&iws[8 + e], 1);
    iws[WS_TOKLIST + iws[16 + e] + p] = t;
}

// ---------------------------------------------------------------------------
// Grouped GEMM: tile 128x128x32, 256 threads = 8 waves (2 M x 4 N),
// each wave does 4x2 WMMA 16x16 tiles. Double-buffered LDS, one barrier/iter.
// ---------------------------------------------------------------------------
template <bool FIRST>
__global__ __launch_bounds__(256) void moe_gemm_kernel(
    const float* __restrict__ Xf,        // FIRST only
    const _Float16* __restrict__ Ah,     // !FIRST only (hbuf)
    const float* __restrict__ W,         // [E][K][N] row-major
    const float* __restrict__ bias,      // [E][N]
    const int* __restrict__ iws,
    _Float16* __restrict__ hbuf,         // FIRST only
    float* __restrict__ out,             // !FIRST only
    int K, int N) {
    const int e  = blockIdx.z;
    const int ne = iws[e];
    const int mt = blockIdx.x;
    if (mt * 128 >= ne) return;
    const int n0  = blockIdx.y * 128;
    const int off = iws[16 + e];
    const int* toklist = iws + WS_TOKLIST;
    const float* Wb = W + (size_t)e * K * N;

    __shared__ __align__(16) _Float16 As[2][128 * LDSTRIDE];
    __shared__ __align__(16) _Float16 Bs[2][128 * LDSTRIDE];
    __shared__ int rowtok[128];

    const int tid = threadIdx.x;
    if (tid < 128) {
        const int r = mt * 128 + tid;
        rowtok[tid] = (r < ne) ? toklist[off + r] : -1;
    }
    __syncthreads();

    // ---- loop-invariant staging descriptors --------------------------------
    // B: each thread owns a 4(k) x 4(n) sub-block -> in-register transpose.
    const int bkb = (tid >> 5) * 4;        // k base (0..28)
    const int bc  = (tid & 31) * 4;        // n base (0..124)
    const float* bsrc[4];
    #pragma unroll
    for (int i = 0; i < 4; ++i)
        bsrc[i] = Wb + (size_t)(bkb + i) * N + n0 + bc;

    // A (FIRST): 4 float4 chunks, gathered rows, mask-out invalid rows.
    const float* asrc[4];
    float amask[4];
    int   adst[4];
    // A (!FIRST): 2 v8h chunks, direct (async) copy of f16 h rows.
    const _Float16* gsrc[2];
    int   gdst[2];
    if (FIRST) {
        #pragma unroll
        for (int j = 0; j < 4; ++j) {
            const int lin = tid + j * 256;
            const int row = lin >> 3;
            const int c   = (lin & 7) * 4;
            const int tok = rowtok[row];
            amask[j] = tok >= 0 ? 1.f : 0.f;
            asrc[j]  = Xf + (size_t)(tok >= 0 ? tok : 0) * DD + c;
            adst[j]  = row * LDSTRIDE + c;
        }
    } else {
        #pragma unroll
        for (int j = 0; j < 2; ++j) {
            const int lin = tid + j * 256;
            const int row = lin >> 2;
            const int c   = (lin & 3) * 8;
            const int r   = mt * 128 + row;
            gsrc[j] = Ah + (size_t)(r < ne ? off + r : off) * K + c;
            gdst[j] = row * LDSTRIDE + c;
        }
    }

    // ---- pipelined staging helpers ----------------------------------------
    v4f breg[4];
    v4f areg[4];
    auto loadB = [&]() {
        #pragma unroll
        for (int i = 0; i < 4; ++i) {
            breg[i] = *(const v4f*)bsrc[i];
            bsrc[i] += (size_t)32 * N;
        }
    };
    auto storeB = [&](int p) {
        #pragma unroll
        for (int i = 0; i < 4; ++i) {  // transpose 4x4: write k-contiguous
            v4h t = {(_Float16)breg[0][i], (_Float16)breg[1][i],
                     (_Float16)breg[2][i], (_Float16)breg[3][i]};
            *(v4h*)&Bs[p][(bc + i) * LDSTRIDE + bkb] = t;
        }
    };
    auto loadA1 = [&]() {
        #pragma unroll
        for (int j = 0; j < 4; ++j) {
            areg[j] = (*(const v4f*)asrc[j]) * amask[j];
            asrc[j] += 32;
        }
    };
    auto storeA1 = [&](int p) {
        #pragma unroll
        for (int j = 0; j < 4; ++j) {
            v4h t = {(_Float16)areg[j][0], (_Float16)areg[j][1],
                     (_Float16)areg[j][2], (_Float16)areg[j][3]};
            *(v4h*)&As[p][adst[j]] = t;
        }
    };
    auto copyA2 = [&](int p) {
        #pragma unroll
        for (int j = 0; j < 2; ++j) {
            async_copy16(gsrc[j], &As[p][gdst[j]]);
            gsrc[j] += 32;
        }
    };

    // ---- fragment geometry (loop-invariant) --------------------------------
    const int lane = tid & 31;
    const int wid  = tid >> 5;
    const int wm   = wid & 1;
    const int wn   = wid >> 1;
    const int lr   = lane & 15;
    const int hk   = (lane >> 4) * 8;
    int aoff[4], boff[2];
    #pragma unroll
    for (int mi = 0; mi < 4; ++mi)
        aoff[mi] = (wm * 64 + mi * 16 + lr) * LDSTRIDE + hk;
    #pragma unroll
    for (int ni = 0; ni < 2; ++ni)
        boff[ni] = (wn * 32 + ni * 16 + lr) * LDSTRIDE + hk;

    v8f acc[4][2];
    #pragma unroll
    for (int mi = 0; mi < 4; ++mi)
        #pragma unroll
        for (int ni = 0; ni < 2; ++ni)
            #pragma unroll
            for (int q = 0; q < 8; ++q) acc[mi][ni][q] = 0.f;

    // ---- prologue: stage tile 0 into buffer 0 ------------------------------
    if (FIRST) { loadA1(); storeA1(0); } else { copyA2(0); }
    loadB();
    storeB(0);
    if (!FIRST) async_wait();

    // ---- main loop: one barrier per k-step ---------------------------------
    const int nk = K / 32;
    int p = 0;
    for (int it = 0; it < nk; ++it) {
        __syncthreads();                     // buf[p] published, buf[p^1] free
        const bool more = (it + 1 < nk);
        if (more) {
            if (!FIRST) copyA2(p ^ 1);       // async straight into other buf
            else        loadA1();            // prefetch to regs
            loadB();
        }

        v16h a[4], b[2];
        #pragma unroll
        for (int mi = 0; mi < 4; ++mi) {
            const _Float16* q = &As[p][aoff[mi]];
            const v8h lo = *(const v8h*)q;
            const v8h hi = *(const v8h*)(q + 16);
            a[mi] = __builtin_shufflevector(lo, hi, 0, 1, 2, 3, 4, 5, 6, 7,
                                            8, 9, 10, 11, 12, 13, 14, 15);
        }
        #pragma unroll
        for (int ni = 0; ni < 2; ++ni) {
            const _Float16* q = &Bs[p][boff[ni]];
            const v8h lo = *(const v8h*)q;
            const v8h hi = *(const v8h*)(q + 16);
            b[ni] = __builtin_shufflevector(lo, hi, 0, 1, 2, 3, 4, 5, 6, 7,
                                            8, 9, 10, 11, 12, 13, 14, 15);
        }
        #pragma unroll
        for (int mi = 0; mi < 4; ++mi)
            #pragma unroll
            for (int ni = 0; ni < 2; ++ni)
                acc[mi][ni] = __builtin_amdgcn_wmma_f32_16x16x32_f16(
                    false, a[mi], false, b[ni], (short)0, acc[mi][ni],
                    false, false);

        if (more) {
            if (FIRST) storeA1(p ^ 1);
            storeB(p ^ 1);
            if (!FIRST) async_wait();
        }
        p ^= 1;
    }

    // ---- epilogue ----------------------------------------------------------
    const float* be = bias + (size_t)e * N;
    #pragma unroll
    for (int mi = 0; mi < 4; ++mi) {
        #pragma unroll
        for (int ni = 0; ni < 2; ++ni) {
            const int n  = n0 + wn * 32 + ni * 16 + lr;
            const float bv = be[n];
            #pragma unroll
            for (int g = 0; g < 8; ++g) {
                const int lrow = wm * 64 + mi * 16 + (lane >> 4) * 8 + g;
                const int r    = mt * 128 + lrow;
                if (r < ne) {
                    float v = acc[mi][ni][g] + bv;
                    if (FIRST) {
                        v = v > 0.f ? v : 0.f;
                        hbuf[(size_t)(off + r) * HH + n] = (_Float16)v;
                    } else {
                        const int tok = rowtok[lrow];
                        out[(size_t)tok * DD + n] = v;
                    }
                }
            }
        }
    }
}

extern "C" void kernel_launch(void* const* d_in, const int* in_sizes, int n_in,
                              void* d_out, int out_size, void* d_ws, size_t ws_size,
                              hipStream_t stream) {
    (void)in_sizes; (void)n_in; (void)out_size; (void)ws_size;
    const float* x  = (const float*)d_in[0];
    const float* Wg = (const float*)d_in[1];
    const float* bg = (const float*)d_in[2];
    const float* W1 = (const float*)d_in[3];
    const float* b1 = (const float*)d_in[4];
    const float* W2 = (const float*)d_in[5];
    const float* b2 = (const float*)d_in[6];
    float* out = (float*)d_out;

    int* iws = (int*)d_ws;
    _Float16* hbuf = (_Float16*)((char*)d_ws + WS_HBUF_BYTE);

    zero_ws_kernel<<<1, 32, 0, stream>>>(iws);
    gate_kernel<<<BT / 8, 256, 0, stream>>>(x, Wg, bg, iws + WS_EXPERT_ID, iws);
    offsets_kernel<<<1, 1, 0, stream>>>(iws);
    scatter_kernel<<<BT / 256, 256, 0, stream>>>(iws);

    moe_gemm_kernel<true><<<dim3(BT / 128, HH / 128, EE), 256, 0, stream>>>(
        x, (const _Float16*)nullptr, W1, b1, iws, hbuf, (float*)nullptr,
        DD, HH);
    moe_gemm_kernel<false><<<dim3(BT / 128, DD / 128, EE), 256, 0, stream>>>(
        (const float*)nullptr, (const _Float16*)hbuf, W2, b2, iws,
        (_Float16*)nullptr, out, HH, DD);
}
